// Mixer_63634235457733
// MI455X (gfx1250) — matrix operand order repaired and checked by reading, not compiled, not run on hardware
//
#include <hip/hip_runtime.h>

#define Bdim 64
#define Cdim 256
#define Ndim 256
#define LDH  264   // padded bf16 row stride (spreads LDS banks for A-fragment loads)

#define ASG __attribute__((address_space(1)))
#define ASL __attribute__((address_space(3)))

typedef __attribute__((ext_vector_type(16))) __bf16        v16bf;
typedef __attribute__((ext_vector_type(2)))  __bf16        v2bf;
typedef __attribute__((ext_vector_type(8)))  float         v8f;
typedef __attribute__((ext_vector_type(4)))  float         v4f;
typedef __attribute__((ext_vector_type(4)))  unsigned int  v4u;
typedef int v4i __attribute__((vector_size(16)));          // matches async builtin param type

#if __has_builtin(__builtin_amdgcn_global_load_async_to_lds_b128) && \
    __has_builtin(__builtin_amdgcn_global_load_async_to_lds_b32)  && \
    __has_builtin(__builtin_amdgcn_s_wait_asynccnt)
#define HAVE_ASYNC_LDS 1
#else
#define HAVE_ASYNC_LDS 0
#endif

// fp32 -> bf16 through native fptrunc (lowers to v_cvt_pk_bf16_f32 on gfx1250)
__device__ __forceinline__ unsigned short f2bf(float f) {
  union { __bf16 h; unsigned short u; } c;
  c.h = (__bf16)f;
  return c.u;
}
__device__ __forceinline__ unsigned int pk_bf16(float a, float b) {
#if __has_builtin(__builtin_amdgcn_cvt_pk_bf16_f32)
  union { v2bf v; unsigned int u; } c;
  c.v = __builtin_amdgcn_cvt_pk_bf16_f32(a, b);
  return c.u;
#else
  union { v2bf v; unsigned int u; } c;
  c.v[0] = (__bf16)a;
  c.v[1] = (__bf16)b;
  return c.u;
#endif
}

// async global -> LDS DMA (ASYNCcnt path); synchronous fallback
__device__ __forceinline__ void async_g2l_b128(const float* gp, float* lp) {
#if HAVE_ASYNC_LDS
  __builtin_amdgcn_global_load_async_to_lds_b128((ASG v4i*)gp, (ASL v4i*)lp, 0, 0);
#else
  *(v4f*)lp = *(const v4f*)gp;
#endif
}
__device__ __forceinline__ void async_g2l_b32(const float* gp, float* lp) {
#if HAVE_ASYNC_LDS
  __builtin_amdgcn_global_load_async_to_lds_b32((ASG int*)gp, (ASL int*)lp, 0, 0);
#else
  *lp = *gp;
#endif
}
__device__ __forceinline__ void wait_async_lds() {
#if HAVE_ASYNC_LDS
  __builtin_amdgcn_s_wait_asynccnt(0);
#endif
}

// branchless GELU (tanh form via v_exp_f32): x * sigmoid(1.59577*(x + 0.044715 x^3))
__device__ __forceinline__ float gelu_f(float x) {
  float u = x * (1.59576912f + 0.07135481f * x * x);
  return x / (1.0f + __expf(-u));
}

union ABu { v16bf v; v4u q[2]; unsigned int u[8]; unsigned short s[16]; };

// D[64 rows (batch) x 256 cols (out)] += A(LDS bf16 [64][LDH], rows x K=256) * W^T
// W: global fp32, row-major [256 out][256 in]. Wave w owns cols [32w, 32w+32).
__device__ __forceinline__ void gemm_64x256(const unsigned short* __restrict__ lact,
                                            const float* __restrict__ W,
                                            int wave, int lane, v8f acc[4][2]) {
  const int lm = lane & 15;
  const int hs = lane >> 4;       // half-wave select
  for (int k0 = 0; k0 < Ndim; k0 += 32) {
    // ---- issue ALL weight loads first (independent regs -> one clause, loads overlap WMMA) ----
    v4f wr[2][4];
#pragma unroll
    for (int c = 0; c < 2; ++c) {
      const float* wp = W + (size_t)(wave * 32 + c * 16 + lm) * Ndim + k0 + hs * 16;
      __builtin_prefetch(wp + 32, 0, 3);    // global_prefetch_b8, near-cache scope
#pragma unroll
      for (int j = 0; j < 4; ++j)
        wr[c][j] = __builtin_nontemporal_load((const v4f*)(wp + 4 * j));
    }
    // ---- A fragments from LDS: 16x32 bf16 ----
    v16bf a[4];
#pragma unroll
    for (int r = 0; r < 4; ++r) {
      const unsigned short* p = lact + (r * 16 + lm) * LDH + k0 + hs * 16;
      ABu u;
      u.q[0] = *(const v4u*)p;
      u.q[1] = *(const v4u*)(p + 8);
      a[r] = u.v;
    }
    // ---- convert weights fp32 -> bf16 (paired cvt) ----
    v16bf b[2];
#pragma unroll
    for (int c = 0; c < 2; ++c) {
      ABu u;
#pragma unroll
      for (int j = 0; j < 4; ++j) {
        u.u[2 * j + 0] = pk_bf16(wr[c][j][0], wr[c][j][1]);
        u.u[2 * j + 1] = pk_bf16(wr[c][j][2], wr[c][j][3]);
      }
      b[c] = u.v;
    }
#pragma unroll
    for (int r = 0; r < 4; ++r)
#pragma unroll
      for (int c = 0; c < 2; ++c)
        acc[r][c] = __builtin_amdgcn_wmma_f32_16x16x32_bf16(
            false, a[r], false, b[c], (short)0, acc[r][c], false, false);
  }
}

// Per-row mean / rsqrt(var+eps) over last dim N for a [rows][N] fp32 tensor.
__global__ __launch_bounds__(256)
void row_stats_kernel(const float* __restrict__ X, float* __restrict__ mu,
                      float* __restrict__ rsv) {
  const int row  = blockIdx.x * 8 + (threadIdx.x >> 5);
  const int lane = threadIdx.x & 31;
  const float* p = X + (size_t)row * Ndim + lane * 8;
  v4f f0 = *(const v4f*)p;
  v4f f1 = *(const v4f*)(p + 4);
  float s = 0.f, s2 = 0.f;
#pragma unroll
  for (int j = 0; j < 4; ++j) {
    s  += f0[j] + f1[j];
    s2 += f0[j] * f0[j] + f1[j] * f1[j];
  }
  for (int off = 16; off > 0; off >>= 1) {
    s  += __shfl_xor(s,  off, 32);
    s2 += __shfl_xor(s2, off, 32);
  }
  if (lane == 0) {
    float m   = s * (1.0f / Ndim);
    float var = s2 * (1.0f / Ndim) - m * m;
    mu[row]  = m;
    rsv[row] = rsqrtf(var + 1e-5f);
  }
}

// One workgroup per instance (channel c in phase0, patch n in phase1).
// slice -(async DMA)-> LDS -> LN(bf16) -> GEMM1 -> +bias,gelu -> LDS -> GEMM2 -> +bias,+residual
__global__ __launch_bounds__(256)
void mixer_phase_kernel(const float* __restrict__ src,
                        const float* __restrict__ mu,  const float* __restrict__ rsv,
                        const float* __restrict__ gam, const float* __restrict__ bet,
                        const float* __restrict__ W1all, const float* __restrict__ B1all,
                        const float* __restrict__ W2all, const float* __restrict__ B2all,
                        float* __restrict__ outp, int phase) {
  extern __shared__ char smem[];
  float*          lx   = (float*)smem;                               // [64][256] fp32 (residual)
  unsigned short* lact = (unsigned short*)(smem + Bdim * Ndim * 4);  // [64][LDH] bf16
  unsigned short* lh   = lact + Bdim * LDH;                          // [64][LDH] bf16

  const int blk  = blockIdx.x;
  const int t    = threadIdx.x;
  const int wave = t >> 5, lane = t & 31;
  const int lm = lane & 15, hs = lane >> 4;

  const float* W1 = W1all + (size_t)blk * Ndim * Ndim;
  const float* W2 = W2all + (size_t)blk * Ndim * Ndim;
  const float* B1 = B1all + (size_t)blk * Ndim;
  const float* B2 = B2all + (size_t)blk * Ndim;

  // ---- stage activation slice into LDS (async DMA path) ----
  if (phase == 0) {                       // x[:, blk, :] is contiguous in n -> b128 bursts
#pragma unroll 4
    for (int it = 0; it < 16; ++it) {
      const int flat4 = it * 256 + t;     // float4 index within [64][64] of float4
      const int row = flat4 >> 6;
      const int col = (flat4 & 63) << 2;
      async_g2l_b128(src + ((size_t)row * Cdim + blk) * Ndim + col, lx + row * Ndim + col);
    }
  } else {                                // xmid[:, :, blk]: stride-N gather, per-element DMA
    for (int r = 0; r < Bdim; ++r)
      async_g2l_b32(src + ((size_t)r * Cdim + t) * Ndim + blk, lx + r * Ndim + t);
  }
  wait_async_lds();
  __syncthreads();

  // ---- layernorm -> bf16 LDS ----
  if (phase == 0) {                       // stats per row b, affine per col n
    const float ge = gam[t], be = bet[t];
    for (int r = 0; r < Bdim; ++r) {
      const int sidx = r * Cdim + blk;
      lact[r * LDH + t] = f2bf((lx[r * Ndim + t] - mu[sidx]) * rsv[sidx] * ge + be);
    }
  } else {                                // stats per element (b, c=t), affine scalar g2[n]
    const float ge = gam[blk], be = bet[blk];
    for (int r = 0; r < Bdim; ++r) {
      const int sidx = r * Cdim + t;
      lact[r * LDH + t] = f2bf((lx[r * Ndim + t] - mu[sidx]) * rsv[sidx] * ge + be);
    }
  }
  __syncthreads();

  // ---- GEMM1 + bias + GELU -> lh ----
  {
    v8f acc[4][2];
    for (int r = 0; r < 4; ++r) for (int c = 0; c < 2; ++c)
      for (int i = 0; i < 8; ++i) acc[r][c][i] = 0.f;
    gemm_64x256(lact, W1, wave, lane, acc);
#pragma unroll
    for (int r = 0; r < 4; ++r)
#pragma unroll
      for (int c = 0; c < 2; ++c) {
        const int col = wave * 32 + c * 16 + lm;
        const float bias = B1[col];
#pragma unroll
        for (int v = 0; v < 8; ++v) {
          const int row = r * 16 + v + hs * 8;
          lh[row * LDH + col] = f2bf(gelu_f(acc[r][c][v] + bias));
        }
      }
  }
  __syncthreads();

  // ---- GEMM2 + bias + residual -> global ----
  {
    v8f acc[4][2];
    for (int r = 0; r < 4; ++r) for (int c = 0; c < 2; ++c)
      for (int i = 0; i < 8; ++i) acc[r][c][i] = 0.f;
    gemm_64x256(lh, W2, wave, lane, acc);
#pragma unroll
    for (int r = 0; r < 4; ++r)
#pragma unroll
      for (int c = 0; c < 2; ++c) {
        const int col = wave * 32 + c * 16 + lm;
        const float bias = B2[col];
#pragma unroll
        for (int v = 0; v < 8; ++v) {
          const int row = r * 16 + v + hs * 8;
          const float val = acc[r][c][v] + bias + lx[row * Ndim + col];
          const size_t a = (phase == 0)
              ? ((size_t)row * Cdim + blk) * Ndim + col
              : ((size_t)row * Cdim + col) * Ndim + blk;
          outp[a] = val;
        }
      }
  }
}

extern "C" void kernel_launch(void* const* d_in, const int* in_sizes, int n_in,
                              void* d_out, int out_size, void* d_ws, size_t ws_size,
                              hipStream_t stream) {
  (void)in_sizes; (void)n_in; (void)out_size; (void)ws_size;
  const float* x   = (const float*)d_in[0];
  const float* g1  = (const float*)d_in[1];
  const float* be1 = (const float*)d_in[2];
  const float* g2  = (const float*)d_in[3];
  const float* be2 = (const float*)d_in[4];
  const float* tw1 = (const float*)d_in[5];
  const float* tb1 = (const float*)d_in[6];
  const float* tw2 = (const float*)d_in[7];
  const float* tb2 = (const float*)d_in[8];
  const float* cw1 = (const float*)d_in[9];
  const float* cb1 = (const float*)d_in[10];
  const float* cw2 = (const float*)d_in[11];
  const float* cb2 = (const float*)d_in[12];
  float* out = (float*)d_out;

  float* xmid = (float*)d_ws;                             // [B,C,N] fp32
  float* mu1  = xmid + (size_t)Bdim * Cdim * Ndim;
  float* rs1  = mu1 + Bdim * Cdim;
  float* mu2  = rs1 + Bdim * Cdim;
  float* rs2  = mu2 + Bdim * Cdim;

  const size_t smem = (size_t)Bdim * Ndim * 4 + 2 * (size_t)Bdim * LDH * 2; // 130 KB

  row_stats_kernel<<<Bdim * Cdim / 8, 256, 0, stream>>>(x, mu1, rs1);
  mixer_phase_kernel<<<Cdim, 256, smem, stream>>>(x, mu1, rs1, g1, be1,
                                                  tw1, tb1, tw2, tb2, xmid, 0);
  row_stats_kernel<<<Bdim * Cdim / 8, 256, 0, stream>>>(xmid, mu2, rs2);
  mixer_phase_kernel<<<Ndim, 256, smem, stream>>>(xmid, mu2, rs2, g2, be2,
                                                  cw1, cb1, cw2, cb2, out, 1);
}
